// simple_onset_frame_48189533061165
// MI455X (gfx1250) — matrix operand order repaired and checked by reading, not compiled, not run on hardware
//
#include <hip/hip_runtime.h>
#include <hip/hip_bf16.h>
#include <math.h>

// ---------------------------------------------------------------------------
// Problem constants (from reference)
// ---------------------------------------------------------------------------
#define N_BATCH   8
#define S_LEN     8192
#define S_TOT     (N_BATCH * S_LEN)   // 65536 positions
#define IN_F      229
#define MC        48                  // model channels
#define OUTF      88                  // onset/frame channels
#define IN2       (MC + OUTF)         // 136
#define QKV_N     (3 * MC)            // 144 (q | k | v)
#define KERNEL_W  31
#define PAD       15
#define GROUPS    8
#define DHEAD     (MC / GROUPS)       // 6

// padded K dims for bf16 WMMA (multiple of 32)
#define KP1 256                       // for K=229
#define KP2 160                       // for K=136

// WMMA vector types (gfx1250, wave32)
typedef __attribute__((ext_vector_type(16))) __bf16 v16bf;
typedef __attribute__((ext_vector_type(8)))  float  v8f;

// ---------------------------------------------------------------------------
// Weight packing: torch Linear [48, KD] x3 -> bf16 B matrix [144][KP],
// row n holds W[n % 48][k] zero-padded to KP.  Row-major over K means each
// lane's WMMA B-fragment (16 consecutive K values for one column n) is a
// single aligned 32-byte load.
// ---------------------------------------------------------------------------
__global__ void pack_weights_kernel(const float* __restrict__ Wq,
                                    const float* __restrict__ Wk,
                                    const float* __restrict__ Wv,
                                    int KD, int KP, __bf16* __restrict__ out) {
    int total = 3 * MC * KP;
    for (int idx = blockIdx.x * blockDim.x + threadIdx.x; idx < total;
         idx += gridDim.x * blockDim.x) {
        int n = idx / KP, k = idx % KP;
        const float* W = (n < MC) ? Wq : (n < 2 * MC ? Wk : Wv);
        float v = (k < KD) ? W[(n % MC) * KD + k] : 0.f;
        out[idx] = (__bf16)v;   // native v_cvt f32->bf16
    }
}

// ---------------------------------------------------------------------------
// Projection GEMM: C[M=65536, 144] = A[M, KD](f32) * Bpack[144, KP]^T(bf16)
// One wave per 16-row M tile, 9 N tiles (=144 cols), fully unrolled K loop of
// v_wmma_f32_16x16x32_bf16 with f32 accumulation.
//
// A-fragment layout (contiguous K order chosen here, matching B; 16-bit A 16x32):
//   lanes 0-15  (M = lane):     e0..15 = K 0..15
//   lanes 16-31 (M = lane-16):  e0..15 = K 16..31
// B-fragment layout (32x16, K-major): lane L -> column N = L%16,
//   16 consecutive K values starting at kk + (L>=16 ? 16 : 0).
// C/D: VGPR r = row (m0 + (L>=16?8:0) + r), col = L%16.
// ---------------------------------------------------------------------------
template <int KD, int KP>
__global__ void proj_gemm_kernel(const float* __restrict__ A,
                                 const __bf16* __restrict__ Bp,
                                 float* __restrict__ C) {
    const int lane = threadIdx.x & 31;
    const int wave = blockIdx.x * (blockDim.x >> 5) + (threadIdx.x >> 5);
    const int m0   = wave * 16;
    const int half = lane >> 4;     // 0 or 1
    const int l15  = lane & 15;

    union { v16bf v; __bf16 h[16]; } afr;
    v8f acc[9];
#pragma unroll
    for (int t = 0; t < 9; ++t)
#pragma unroll
        for (int j = 0; j < 8; ++j) acc[t][j] = 0.f;

    const float* __restrict__ arow = A + (long)(m0 + l15) * KD;

#pragma unroll
    for (int kk = 0; kk < KP; kk += 32) {
        const int kbA = kk + half * 16;
        if (kk + 32 <= KD) {
            // compile-time full iteration: unconditional loads
#pragma unroll
            for (int j = 0; j < 8; ++j) {
                afr.h[j]     = (__bf16)arow[kbA + j];
                afr.h[8 + j] = (__bf16)arow[kbA + 8 + j];
            }
        } else {
            // K tail: branchless (clamped-address load + select), no exec div
#pragma unroll
            for (int j = 0; j < 8; ++j) {
                int k0 = kbA + j;
                int k1 = kbA + 8 + j;
                float v0 = arow[k0 < KD ? k0 : 0];
                float v1 = arow[k1 < KD ? k1 : 0];
                afr.h[j]     = (__bf16)(k0 < KD ? v0 : 0.f);
                afr.h[8 + j] = (__bf16)(k1 < KD ? v1 : 0.f);
            }
        }
        const int kbB = kk + half * 16;
#pragma unroll
        for (int nt = 0; nt < 9; ++nt) {
            int n = nt * 16 + l15;
            v16bf bfr = *reinterpret_cast<const v16bf*>(Bp + n * KP + kbB); // 32B aligned
            acc[nt] = __builtin_amdgcn_wmma_f32_16x16x32_bf16(
                false, afr.v, false, bfr, (short)0, acc[nt], false, false);
        }
    }

    const int mrow = m0 + half * 8;
#pragma unroll
    for (int nt = 0; nt < 9; ++nt)
#pragma unroll
        for (int r = 0; r < 8; ++r)
            C[(long)(mrow + r) * QKV_N + nt * 16 + l15] = acc[nt][r];
}

// ---------------------------------------------------------------------------
// Fused sliding-window attention + LayerNorm + sigmoid head (+ concat).
// One workgroup = 64 positions of one batch; k/v window staged in LDS.
// ---------------------------------------------------------------------------
#define CHUNK 64
#define WIN   (CHUNK + 2 * PAD)   // 94

__global__ void attn_block_kernel(const float* __restrict__ qkv,     // [S_TOT][144]
                                  const float* __restrict__ rel,     // [48][31]
                                  const float* __restrict__ lng,
                                  const float* __restrict__ lnb,
                                  const float* __restrict__ Wp,      // [88][48]
                                  const float* __restrict__ bp,      // [88]
                                  float* __restrict__ pred,          // [S_TOT][88]
                                  float* __restrict__ attn_out,      // [S_TOT][8][31] or null
                                  float* __restrict__ concat_out) {  // [S_TOT][136] or null
    __shared__ float sk[WIN * MC];
    __shared__ float sv[WIN * MC];
    __shared__ float srel[MC * KERNEL_W];
    __shared__ float sW[OUTF * MC];
    __shared__ float sb[OUTF];
    __shared__ float sg[MC];
    __shared__ float sbeta[MC];

    const int tid   = threadIdx.x;
    const int nchnk = S_LEN / CHUNK;
    const int batch = blockIdx.x / nchnk;
    const int chunk = blockIdx.x % nchnk;
    const int p0    = chunk * CHUNK;
    const long base = (long)batch * S_LEN;

    // stage k/v window (zero beyond sequence ends), rel, head weights
    for (int i = tid; i < WIN * 2 * MC; i += CHUNK) {
        int pos = i / (2 * MC), c = i % (2 * MC);
        int p = p0 - PAD + pos;
        float v = 0.f;
        if (p >= 0 && p < S_LEN) v = qkv[(base + p) * QKV_N + MC + c];
        if (c < MC) sk[pos * MC + c] = v;
        else        sv[pos * MC + (c - MC)] = v;
    }
    for (int i = tid; i < MC * KERNEL_W; i += CHUNK) srel[i] = rel[i];
    for (int i = tid; i < OUTF * MC;     i += CHUNK) sW[i]   = Wp[i];
    for (int i = tid; i < OUTF;          i += CHUNK) sb[i]   = bp[i];
    for (int i = tid; i < MC; i += CHUNK) { sg[i] = lng[i]; sbeta[i] = lnb[i]; }
    __syncthreads();

    const long gpos = base + p0 + tid;

    float q[MC];
#pragma unroll
    for (int c = 0; c < MC; ++c) q[c] = qkv[gpos * QKV_N + c];

    float xo[MC];
    for (int g = 0; g < GROUPS; ++g) {
        float e[KERNEL_W];
        float mx = -1e30f;
#pragma unroll
        for (int kp = 0; kp < KERNEL_W; ++kp) {
            float s = 0.f;
#pragma unroll
            for (int d = 0; d < DHEAD; ++d) {
                int c = g * DHEAD + d;
                s += q[c] * (sk[(tid + kp) * MC + c] + srel[c * KERNEL_W + kp]);
            }
            e[kp] = s;
            mx = fmaxf(mx, s);
        }
        float den = 0.f;
#pragma unroll
        for (int kp = 0; kp < KERNEL_W; ++kp) { e[kp] = __expf(e[kp] - mx); den += e[kp]; }
        float inv = 1.f / den;
#pragma unroll
        for (int kp = 0; kp < KERNEL_W; ++kp) e[kp] *= inv;

        if (attn_out) {
#pragma unroll
            for (int kp = 0; kp < KERNEL_W; ++kp)
                attn_out[(gpos * GROUPS + g) * KERNEL_W + kp] = e[kp];
        }
#pragma unroll
        for (int d = 0; d < DHEAD; ++d) {
            int c = g * DHEAD + d;
            float a = 0.f;
#pragma unroll
            for (int kp = 0; kp < KERNEL_W; ++kp) a += e[kp] * sv[(tid + kp) * MC + c];
            xo[c] = a;
        }
    }

    // LayerNorm over 48 channels
    float mu = 0.f;
#pragma unroll
    for (int c = 0; c < MC; ++c) mu += xo[c];
    mu *= (1.f / MC);
    float var = 0.f;
#pragma unroll
    for (int c = 0; c < MC; ++c) { float d = xo[c] - mu; var += d * d; }
    var *= (1.f / MC);
    float rstd = rsqrtf(var + 1e-5f);
#pragma unroll
    for (int c = 0; c < MC; ++c) xo[c] = (xo[c] - mu) * rstd * sg[c] + sbeta[c];

    // sigmoid head (48 -> 88)
    for (int o = 0; o < OUTF; ++o) {
        float s = sb[o];
#pragma unroll
        for (int c = 0; c < MC; ++c) s += xo[c] * sW[o * MC + c];
        float sig = 1.f / (1.f + __expf(-s));
        pred[gpos * OUTF + o] = sig;
        if (concat_out) concat_out[gpos * IN2 + o] = sig;
    }
    if (concat_out) {
#pragma unroll
        for (int c = 0; c < MC; ++c) concat_out[gpos * IN2 + OUTF + c] = xo[c];
    }
}

// ---------------------------------------------------------------------------
// Launch: pack -> gemm1 -> attn1(fused onset head + concat) -> gemm2 -> attn2
// ---------------------------------------------------------------------------
extern "C" void kernel_launch(void* const* d_in, const int* in_sizes, int n_in,
                              void* d_out, int out_size, void* d_ws, size_t ws_size,
                              hipStream_t stream) {
    const float* spec = (const float*)d_in[0];
    const float* Wq1  = (const float*)d_in[1];
    const float* Wk1  = (const float*)d_in[2];
    const float* Wv1  = (const float*)d_in[3];
    const float* rel1 = (const float*)d_in[4];
    const float* ln1g = (const float*)d_in[5];
    const float* ln1b = (const float*)d_in[6];
    const float* Wo   = (const float*)d_in[7];
    const float* bo   = (const float*)d_in[8];
    const float* Wq2  = (const float*)d_in[9];
    const float* Wk2  = (const float*)d_in[10];
    const float* Wv2  = (const float*)d_in[11];
    const float* rel2 = (const float*)d_in[12];
    const float* ln2g = (const float*)d_in[13];
    const float* ln2b = (const float*)d_in[14];
    const float* Wf   = (const float*)d_in[15];
    const float* bf   = (const float*)d_in[16];

    // workspace layout (bytes)
    const size_t OFF_B1  = 0;                                   // 144*256*2
    const size_t OFF_B2  = 131072;                              // 144*160*2
    const size_t OFF_QKV = 262144;                              // 65536*144*4
    const size_t OFF_X2  = OFF_QKV + (size_t)S_TOT * QKV_N * 4; // 65536*136*4

    __bf16* B1 = (__bf16*)((char*)d_ws + OFF_B1);
    __bf16* B2 = (__bf16*)((char*)d_ws + OFF_B2);
    float* qkv = (float*)((char*)d_ws + OFF_QKV);
    float* X2  = (float*)((char*)d_ws + OFF_X2);

    float* out   = (float*)d_out;
    float* frame = out;                              // [8,8192,88]
    float* onset = out + (long)S_TOT * OUTF;         // [8,8192,88]
    float* attn  = out + 2L * S_TOT * OUTF;          // [8,8192,8,31]

    pack_weights_kernel<<<64, 256, 0, stream>>>(Wq1, Wk1, Wv1, IN_F, KP1, B1);
    pack_weights_kernel<<<64, 256, 0, stream>>>(Wq2, Wk2, Wv2, IN2,  KP2, B2);

    // 4096 M-tiles, 4 waves (128 threads) per block
    proj_gemm_kernel<IN_F, KP1><<<S_TOT / 16 / 4, 128, 0, stream>>>(spec, B1, qkv);

    attn_block_kernel<<<S_TOT / CHUNK, CHUNK, 0, stream>>>(
        qkv, rel1, ln1g, ln1b, Wo, bo, onset, attn, X2);

    proj_gemm_kernel<IN2, KP2><<<S_TOT / 16 / 4, 128, 0, stream>>>(X2, B2, qkv);

    attn_block_kernel<<<S_TOT / CHUNK, CHUNK, 0, stream>>>(
        qkv, rel2, ln2g, ln2b, Wf, bf, frame, nullptr, nullptr);
}
